// GNNEncoder_7473243095123
// MI455X (gfx1250) — matrix operand order repaired and checked by reading, not compiled, not run on hardware
//
#include <hip/hip_runtime.h>
#include <hip/hip_bf16.h>

// ============================================================================
// GNNEncoder forward for MI455X (gfx1250, wave32, WMMA).
//  H=128, B=4, V=256, 3 layers, OUT_CH=2.
//  All 128x128 GEMMs run as v_wmma_f32_16x16x32_f16 (f16 in, f32 accumulate).
//  Weights are re-packed once per launch into WMMA B-fragment order (f16).
//  LayerNorm / sigmoid / silu are evaluated directly on the WMMA C-fragment
//  register layout using 16-lane shfl_xor row reductions (wave32 halves).
//  All cross-wave reductions are deterministic (fixed-order partial sums).
//  Small-vector staging uses GLOBAL_LOAD_ASYNC_TO_LDS_B32 (ASYNCcnt path);
//  Ah/Vh row blocks are warmed with global_prefetch_b8.
//
// Assumed d_in flattening (depth-first, dict insertion order, w before b):
//  [0] x (B,V,2)  [1] graph (B,V,V)  [2] timesteps (B,) int32
//  [3] edge_embed.w [4] edge_embed.b [5] time1.w [6] time1.b [7] time2.w [8] time2.b
//  per layer l (base 9+20l): U.w U.b V.w V.b A.w A.b B.w B.b C.w C.b
//                            norm_h.g norm_h.b norm_e.g norm_e.b time.w time.b
//                            plo_ln.g plo_ln.b plo_lin.w plo_lin.b
//  [69] out.gn_g [70] out.gn_b [71] out.conv.w [72] out.conv.b
// ============================================================================

typedef _Float16 v16h __attribute__((ext_vector_type(16)));
typedef _Float16 v4h  __attribute__((ext_vector_type(4)));
typedef float    v8f  __attribute__((ext_vector_type(8)));

#define HF   128            // hidden
#define VV   256            // nodes
#define NB   4              // batch
#define NROW (NB*VV*VV)     // 262144 edge rows
#define HROW (NB*VV)        // 1024 node rows
#define MATH (8*4*32*16)    // 16384 halves per packed 128x128 matrix
#define NMAT 19             // packed matrices: edge + 6 per layer * 3

// ---------------------------------------------------------------- WMMA helpers
__device__ __forceinline__ v8f wmma_f16(v16h a, v16h b, v8f c) {
  // (neg_a, A, neg_b, B, c_mod, C, reuse_a, reuse_b)
  return __builtin_amdgcn_wmma_f32_16x16x32_f16(false, a, false, b, (short)0, c,
                                                false, false);
}

// Packed B fragment: layout [ct][kc][lane][16 halves], 32B per lane -> 2xb128.
__device__ __forceinline__ v16h bfrag(const _Float16* __restrict__ pk,
                                      int ct, int kc, int lane) {
  return ((const v16h*)pk)[(ct*4 + kc)*32 + lane];
}

// A fragment from an LDS row-major f16 tile [16][128].
// ISA 7.12.2: lane<16 holds M=lane, K = 32kc + {0..7,16..23};
//             lane>=16 holds M=lane-16, K = 32kc + {8..15,24..31}.
__device__ __forceinline__ v16h afrag(const _Float16* lds, int lane, int kc) {
  int m = lane & 15;
  int basek = (lane >> 4) * 8;
  const _Float16* row = lds + m*HF;
  v16h a;
#pragma unroll
  for (int jj = 0; jj < 8; ++jj) {
    int k = 32*kc + basek + 2*jj + (jj >= 4 ? 8 : 0);
    a[2*jj]   = row[k];
    a[2*jj+1] = row[k+1];
  }
  return a;
}

// Coalesced stage of 16 rows x 128 f32 -> f16 row-major LDS tile.
__device__ __forceinline__ void stage16(const float* __restrict__ src,
                                        _Float16* lds, int lane) {
  const float4* p = (const float4*)src;
#pragma unroll
  for (int it = 0; it < 16; ++it) {
    int idx = lane + 32*it;                 // 512 float4s
    float4 v = p[idx];
    v4h hv = { (_Float16)v.x, (_Float16)v.y, (_Float16)v.z, (_Float16)v.w };
    *(v4h*)(lds + idx*4) = hv;
  }
}

// One wave: 16x128 GEMM tile (K=128), bias init, f32 store.
__device__ __forceinline__ void gemm_store(const v16h a[4],
                                           const _Float16* __restrict__ pk,
                                           const float* __restrict__ bias,
                                           float* __restrict__ out, int lane) {
  int nloc = lane & 15, halfId = lane >> 4;
#pragma unroll
  for (int ct = 0; ct < 8; ++ct) {
    int n = 16*ct + nloc;
    float bi = bias[n];
    v8f c;
#pragma unroll
    for (int r = 0; r < 8; ++r) c[r] = bi;
#pragma unroll
    for (int kc = 0; kc < 4; ++kc) c = wmma_f16(a[kc], bfrag(pk, ct, kc, lane), c);
#pragma unroll
    for (int r = 0; r < 8; ++r) out[(r + 8*halfId)*HF + n] = c[r];
  }
}

// ---------------------------------------------------------------- k_pack
struct PackArgs { const float* src[NMAT]; };

__global__ __launch_bounds__(256) void k_pack(PackArgs A, _Float16* __restrict__ dst) {
  int idx = blockIdx.x*256 + threadIdx.x;           // < NMAT*MATH
  if (idx >= NMAT*MATH) return;
  int mat = idx >> 14;                              // 16384 per matrix
  int r   = idx & 16383;
  int j    = r & 15;
  int lane = (r >> 4) & 31;
  int kc   = (r >> 9) & 3;
  int ct   = r >> 11;
  int n = 16*ct + (lane & 15);
  int k = 32*kc + (lane >> 4)*8 + (j & 7) + (j >> 3)*16; // B: lane<16 K={0..7,16..23}, else {8..15,24..31}
  dst[idx] = (_Float16)A.src[mat][k*HF + n];
}

// ---------------------------------------------------------------- k_h0 (pos embed)
__global__ __launch_bounds__(256) void k_h0(const float* __restrict__ x,
                                            float* __restrict__ h) {
  int idx = blockIdx.x*256 + threadIdx.x;           // < 131072
  int row = idx >> 7, f = idx & 127;
  float val = (f < 64) ? x[row*2 + 0] : x[row*2 + 1];   // py then px
  int fl = f & 63;
  int pr = fl >> 1;
  float ang = val * 6.2831853071795864f;            // * 2*pi
  float a = ang * exp2f(-(float)pr * 0.4152410118609203f); // /10000^(pr/32)
  h[idx] = (fl & 1) ? cosf(a) : sinf(a);
}

// ---------------------------------------------------------------- k_temb
struct TembArgs {
  const int* ts;
  const float *t1W, *t1B, *t2W, *t2B;
  const float *tW0, *tB0, *tW1, *tB1, *tW2, *tB2;
  float* tvec;                                      // [3][NB][HF]
};

__global__ __launch_bounds__(128) void k_temb(TembArgs A) {
  __shared__ float temb[NB][128], t1[NB][64], rt[NB][64];
  int t = threadIdx.x;                              // 128 threads
  for (int b = 0; b < NB; ++b) {
    float tv = (float)A.ts[b];
    float r;
    if (t < 64) { float fr = expf(-9.210340371976184f * (float)t / 64.f);
                  r = cosf(tv * fr); }
    else        { int k = t - 64;
                  float fr = expf(-9.210340371976184f * (float)k / 64.f);
                  r = sinf(tv * fr); }
    temb[b][t] = r;
  }
  __syncthreads();
  for (int idx = t; idx < NB*64; idx += 128) {      // time1 + relu
    int b = idx >> 6, o = idx & 63;
    float s = A.t1B[o];
    for (int k = 0; k < 128; ++k) s += temb[b][k] * A.t1W[k*64 + o];
    t1[b][o] = fmaxf(s, 0.f);
  }
  __syncthreads();
  for (int idx = t; idx < NB*64; idx += 128) {      // time2, then relu for layers
    int b = idx >> 6, o = idx & 63;
    float s = A.t2B[o];
    for (int k = 0; k < 64; ++k) s += t1[b][k] * A.t2W[k*64 + o];
    rt[b][o] = fmaxf(s, 0.f);
  }
  __syncthreads();
  const float* Ws[3] = { A.tW0, A.tW1, A.tW2 };
  const float* Bs[3] = { A.tB0, A.tB1, A.tB2 };
  for (int l = 0; l < 3; ++l)
    for (int idx = t; idx < NB*HF; idx += 128) {
      int b = idx >> 7, o = idx & 127;
      float s = Bs[l][o];
      for (int k = 0; k < 64; ++k) s += rt[b][k] * Ws[l][k*HF + o];
      A.tvec[(l*NB + b)*HF + o] = s;
    }
}

// ---------------------------------------------------------------- k_embed
// e0 = scalar_embed_sine(graph) @ We + be, sincos built directly in A-fragment
// layout per lane (no intermediate tensor).
__global__ __launch_bounds__(256) void k_embed(const float* __restrict__ graph,
                                               const _Float16* __restrict__ pE,
                                               const float* __restrict__ bE,
                                               float* __restrict__ e) {
  int t = threadIdx.x, w = t >> 5, lane = t & 31;
  int rowbase = blockIdx.x*128 + w*16;
  int m = lane & 15, halfId = lane >> 4;
  float g = graph[rowbase + m];
  v16h a[4];
#pragma unroll
  for (int kc = 0; kc < 4; ++kc) {
#pragma unroll
    for (int jj = 0; jj < 8; ++jj) {
      int p = 16*kc + 4*halfId + jj + (jj >= 4 ? 4 : 0);   // feature pair index
      float ang = g * exp2f(-(float)p * 0.2076205059304601f); // /10000^(p/64)
      a[kc][2*jj]   = (_Float16)sinf(ang);
      a[kc][2*jj+1] = (_Float16)cosf(ang);
    }
  }
  gemm_store(a, pE, bE, e + (size_t)rowbase*HF, lane);
}

// ---------------------------------------------------------------- k_nodes
__global__ __launch_bounds__(256) void k_nodes(const float* __restrict__ h,
    const _Float16* pU, const _Float16* pV, const _Float16* pA, const _Float16* pBm,
    const float* bU, const float* bV, const float* bA, const float* bBv,
    float* Uh, float* Vh, float* Ah, float* Bh) {
  __shared__ _Float16 sH[8][16*HF];
  int t = threadIdx.x, w = t >> 5, lane = t & 31;
  int row0 = (blockIdx.x*8 + w) * 16;
  stage16(h + (size_t)row0*HF, sH[w], lane);
  v16h a[4];
#pragma unroll
  for (int kc = 0; kc < 4; ++kc) a[kc] = afrag(sH[w], lane, kc);
  gemm_store(a, pU,  bU,  Uh + (size_t)row0*HF, lane);
  gemm_store(a, pV,  bV,  Vh + (size_t)row0*HF, lane);
  gemm_store(a, pA,  bA,  Ah + (size_t)row0*HF, lane);
  gemm_store(a, pBm, bBv, Bh + (size_t)row0*HF, lane);
}

// ---------------------------------------------------------------- k_edge
// One pass over e (in place): C-GEMM + Ah[j] + Bh[i] -> gates/agg partials ->
// LN(norm_e)+relu+time -> LN(plo_ln)+silu -> plo-GEMM -> e += plo.
__global__ __launch_bounds__(256) void k_edge(float* __restrict__ e,
    const float* __restrict__ Ah, const float* __restrict__ Bh,
    const float* __restrict__ Vh,
    const _Float16* __restrict__ packC, const _Float16* __restrict__ packP,
    const float* __restrict__ cB, const float* __restrict__ pB,
    const float* __restrict__ neG, const float* __restrict__ neB,
    const float* __restrict__ plG, const float* __restrict__ plB,
    const float* __restrict__ tvecL, float* __restrict__ aggPart) {
  __shared__ float sVec[8][HF];
  __shared__ float sPart[8][HF];
  __shared__ _Float16 sBuf[8][16*HF];                 // reused: e-f16, then silu-f16

  int t = threadIdx.x, w = t >> 5, lane = t & 31;
  int R0 = blockIdx.x*128;
  int b = R0 >> 16, i = (R0 >> 8) & 255;

  // Warm the Ah/Vh row blocks this CTA will gather from (global_prefetch_b8).
  {
    size_t jblk = ((size_t)b*VV + (R0 & 255))*HF;     // 128 rows * 512B
    __builtin_prefetch(Ah + jblk + (size_t)t*64, 0, 3);
    __builtin_prefetch(Vh + jblk + (size_t)t*64, 0, 3);
  }

  // Stage small vectors straight into LDS via async loads (ASYNCcnt path).
  for (int idx = t; idx < 8*HF; idx += 256) {
    int vsel = idx >> 7, c = idx & 127;
    const float* s;
    switch (vsel) {
      case 0: s = cB;  break;
      case 1: s = neG; break;
      case 2: s = neB; break;
      case 3: s = plG; break;
      case 4: s = plB; break;
      case 5: s = pB;  break;
      case 6: s = tvecL + b*HF; break;
      default: s = Bh + ((size_t)b*VV + i)*HF; break;
    }
    unsigned lds_off = (unsigned)(size_t)(&sVec[vsel][c]);
    const float* ga = s + c;
    asm volatile("global_load_async_to_lds_b32 %0, %1, off"
                 :: "v"(lds_off), "v"(ga) : "memory");
  }

  int rowbase = R0 + w*16;
  int j0w = rowbase & 255;
  int nloc = lane & 15, halfId = lane >> 4;

  // ---- GEMM1: en = e @ Wc + bc  (stage A tile while async loads fly)
  stage16(e + (size_t)rowbase*HF, sBuf[w], lane);
  v16h a[4];
#pragma unroll
  for (int kc = 0; kc < 4; ++kc) a[kc] = afrag(sBuf[w], lane, kc);

  asm volatile("s_wait_asynccnt 0" ::: "memory");     // sVec resident in LDS
  __syncthreads();

  v8f acc[8];
#pragma unroll
  for (int ct = 0; ct < 8; ++ct) {
    int n = 16*ct + nloc;
    float bi = sVec[0][n];
    v8f c;
#pragma unroll
    for (int r = 0; r < 8; ++r) c[r] = bi;
#pragma unroll
    for (int kc = 0; kc < 4; ++kc) c = wmma_f16(a[kc], bfrag(packC, ct, kc, lane), c);
    acc[ct] = c;
  }

  // ---- en += Ah[b,j] + Bh[b,i]; gates; agg partials (sum over tile rows j)
  float pa[8];
#pragma unroll
  for (int ct = 0; ct < 8; ++ct) pa[ct] = 0.f;
#pragma unroll
  for (int ct = 0; ct < 8; ++ct) {
    int n = 16*ct + nloc;
#pragma unroll
    for (int r = 0; r < 8; ++r) {
      int M = r + 8*halfId;
      int j = j0w + M;
      float en = acc[ct][r] + Ah[((size_t)b*VV + j)*HF + n] + sVec[7][n];
      acc[ct][r] = en;
      float gt = 1.f / (1.f + __expf(-en));
      pa[ct] += gt * Vh[((size_t)b*VV + j)*HF + n];
    }
  }
#pragma unroll
  for (int ct = 0; ct < 8; ++ct) pa[ct] += __shfl_xor(pa[ct], 16, 32);
  if (lane < 16) {
#pragma unroll
    for (int ct = 0; ct < 8; ++ct) sPart[w][16*ct + lane] = pa[ct];
  }
  __syncthreads();
  if (t < HF) {                                       // deterministic block partial
    float s = 0.f;
#pragma unroll
    for (int w2 = 0; w2 < 8; ++w2) s += sPart[w2][t];
    int halfblk = (R0 >> 7) & 1;
    aggPart[(((size_t)b*VV + i)*2 + halfblk)*HF + t] = s;
  }

  // ---- LN(norm_e) per row (stats via 16-lane shfl within wave32 half)
  float ps[8], pq[8], mean[8], inv[8];
#pragma unroll
  for (int r = 0; r < 8; ++r) { ps[r] = 0.f; pq[r] = 0.f; }
#pragma unroll
  for (int ct = 0; ct < 8; ++ct)
#pragma unroll
    for (int r = 0; r < 8; ++r) { float v = acc[ct][r]; ps[r] += v; pq[r] += v*v; }
#pragma unroll
  for (int r = 0; r < 8; ++r) {
#pragma unroll
    for (int off = 1; off < 16; off <<= 1) {
      ps[r] += __shfl_xor(ps[r], off, 32);
      pq[r] += __shfl_xor(pq[r], off, 32);
    }
    mean[r] = ps[r] * (1.f/128.f);
    float var = pq[r] * (1.f/128.f) - mean[r]*mean[r];
    inv[r] = rsqrtf(var + 1e-5f);
  }
#pragma unroll
  for (int ct = 0; ct < 8; ++ct) {
    int n = 16*ct + nloc;
#pragma unroll
    for (int r = 0; r < 8; ++r) {
      float vv = (acc[ct][r] - mean[r]) * inv[r] * sVec[1][n] + sVec[2][n];
      acc[ct][r] = fmaxf(vv, 0.f) + sVec[6][n];       // relu + time conditioning
    }
  }

  // ---- LN(plo_ln) + silu -> sBuf (f16, A layout via LDS transpose)
#pragma unroll
  for (int r = 0; r < 8; ++r) { ps[r] = 0.f; pq[r] = 0.f; }
#pragma unroll
  for (int ct = 0; ct < 8; ++ct)
#pragma unroll
    for (int r = 0; r < 8; ++r) { float v = acc[ct][r]; ps[r] += v; pq[r] += v*v; }
#pragma unroll
  for (int r = 0; r < 8; ++r) {
#pragma unroll
    for (int off = 1; off < 16; off <<= 1) {
      ps[r] += __shfl_xor(ps[r], off, 32);
      pq[r] += __shfl_xor(pq[r], off, 32);
    }
    mean[r] = ps[r] * (1.f/128.f);
    float var = pq[r] * (1.f/128.f) - mean[r]*mean[r];
    inv[r] = rsqrtf(var + 1e-5f);
  }
#pragma unroll
  for (int ct = 0; ct < 8; ++ct) {
    int n = 16*ct + nloc;
#pragma unroll
    for (int r = 0; r < 8; ++r) {
      float nv = (acc[ct][r] - mean[r]) * inv[r] * sVec[3][n] + sVec[4][n];
      float sv = nv / (1.f + __expf(-nv));            // silu
      int M = r + 8*halfId;
      sBuf[w][M*HF + n] = (_Float16)sv;
    }
  }

  // ---- GEMM2: plo = s @ Wplo + bplo ; e += plo (residual, in place)
  v16h a2[4];
#pragma unroll
  for (int kc = 0; kc < 4; ++kc) a2[kc] = afrag(sBuf[w], lane, kc);
#pragma unroll
  for (int ct = 0; ct < 8; ++ct) {
    int n = 16*ct + nloc;
    float bi = sVec[5][n];
    v8f c;
#pragma unroll
    for (int r = 0; r < 8; ++r) c[r] = bi;
#pragma unroll
    for (int kc = 0; kc < 4; ++kc) c = wmma_f16(a2[kc], bfrag(packP, ct, kc, lane), c);
#pragma unroll
    for (int r = 0; r < 8; ++r) {
      int M = r + 8*halfId;
      size_t idx = ((size_t)rowbase + M)*HF + n;
      e[idx] = e[idx] + c[r];
    }
  }
}

// ---------------------------------------------------------------- k_hupdate
__global__ __launch_bounds__(128) void k_hupdate(float* __restrict__ h,
    const float* __restrict__ Uh, const float* __restrict__ aggPart,
    const float* __restrict__ gH, const float* __restrict__ bH) {
  __shared__ float sh[128], sq[128];
  int row = blockIdx.x, t = threadIdx.x;
  float u = Uh[(size_t)row*HF + t]
          + aggPart[((size_t)row*2 + 0)*HF + t]
          + aggPart[((size_t)row*2 + 1)*HF + t];
  sh[t] = u; sq[t] = u*u;
  __syncthreads();
  for (int off = 64; off > 0; off >>= 1) {
    if (t < off) { sh[t] += sh[t+off]; sq[t] += sq[t+off]; }
    __syncthreads();
  }
  float mean = sh[0] * (1.f/128.f);
  float var  = sq[0] * (1.f/128.f) - mean*mean;
  float hn = (u - mean) * rsqrtf(var + 1e-5f) * gH[t] + bH[t];
  h[(size_t)row*HF + t] += fmaxf(hn, 0.f);
}

// ---------------------------------------------------------------- k_gnstats
__global__ __launch_bounds__(256) void k_gnstats(const float* __restrict__ e,
    float* __restrict__ gPS, float* __restrict__ gPQ) {
  __shared__ float ss[256], sq[256];
  size_t r0 = (size_t)blockIdx.x * 256;
  int t = threadIdx.x, c = t & 127, rr = t >> 7;
  int b = (int)(r0 >> 16);
  (void)b;
  float s = 0.f, q = 0.f;
  for (int r = rr; r < 256; r += 2) {
    float v = e[(r0 + r)*HF + c];
    s += v; q += v*v;
  }
  ss[t] = s; sq[t] = q;
  __syncthreads();
  if (t < 32) {                                       // 32 groups of 4 channels
    float as = 0.f, aq = 0.f;
    for (int cc = 4*t; cc < 4*t + 4; ++cc) { as += ss[cc] + ss[cc+128];
                                             aq += sq[cc] + sq[cc+128]; }
    gPS[(size_t)blockIdx.x*32 + t] = as;              // (b is implied by block)
    gPQ[(size_t)blockIdx.x*32 + t] = aq;
  }
}

// ---------------------------------------------------------------- k_out
__global__ __launch_bounds__(256) void k_out(const float* __restrict__ e,
    const float* __restrict__ gG, const float* __restrict__ gB,
    const float* __restrict__ cW, const float* __restrict__ cBs,
    const float* __restrict__ gPS, const float* __restrict__ gPQ,
    float* __restrict__ out) {
  __shared__ float sW[256], sG[128], sB[128], sM[128], sI[128], sCB[2];
  size_t r0 = (size_t)blockIdx.x * 256;
  int t = threadIdx.x;
  int b = (int)(r0 >> 16);
  if (t < 128) {
    sG[t] = gG[t]; sB[t] = gB[t];
    int g = t >> 2;
    float s = 0.f, q = 0.f;                           // deterministic partial sum
    for (int kb = 0; kb < 256; ++kb) {
      s += gPS[((size_t)b*256 + kb)*32 + g];
      q += gPQ[((size_t)b*256 + kb)*32 + g];
    }
    const float N = 262144.f;                          // 4 * 256 * 256
    float m = s / N, v = q / N - m*m;
    sM[t] = m; sI[t] = rsqrtf(v + 1e-5f);
  }
  sW[t] = cW[t];
  if (t < 2) sCB[t] = cBs[t];
  __syncthreads();
  size_t row = r0 + t;
  int i = (int)((row >> 8) & 255), j = (int)(row & 255);
  const float4* er = (const float4*)(e + row*HF);
  float a0 = sCB[0], a1 = sCB[1];
  for (int c4 = 0; c4 < 32; ++c4) {
    float4 v4 = er[c4];
    float vv[4] = { v4.x, v4.y, v4.z, v4.w };
#pragma unroll
    for (int k = 0; k < 4; ++k) {
      int c = 4*c4 + k;
      float nv = fmaxf((vv[k] - sM[c]) * sI[c] * sG[c] + sB[c], 0.f);
      a0 += nv * sW[c*2 + 0];
      a1 += nv * sW[c*2 + 1];
    }
  }
  out[(((size_t)b*2 + 0)*VV + i)*VV + j] = a0;
  out[(((size_t)b*2 + 1)*VV + i)*VV + j] = a1;
}

// ============================================================================
extern "C" void kernel_launch(void* const* d_in, const int* in_sizes, int n_in,
                              void* d_out, int out_size, void* d_ws, size_t ws_size,
                              hipStream_t stream) {
  (void)in_sizes; (void)n_in; (void)out_size; (void)ws_size;
  const float* x     = (const float*)d_in[0];
  const float* graph = (const float*)d_in[1];
  const int*   ts    = (const int*)d_in[2];
  int p = 3;
  auto F = [&](int i) { return (const float*)d_in[i]; };
  const float *eeW = F(p++), *eeB = F(p++);
  const float *t1W = F(p++), *t1B = F(p++), *t2W = F(p++), *t2B = F(p++);
  struct LayerP {
    const float *uW,*uB,*vW,*vB,*aW,*aB,*bW,*bB,*cW,*cB;
    const float *nhG,*nhB,*neG,*neB,*tW,*tB,*plG,*plB,*pW,*pB;
  } L[3];
  for (int l = 0; l < 3; ++l) {
    L[l].uW  = F(p++); L[l].uB  = F(p++);
    L[l].vW  = F(p++); L[l].vB  = F(p++);
    L[l].aW  = F(p++); L[l].aB  = F(p++);
    L[l].bW  = F(p++); L[l].bB  = F(p++);
    L[l].cW  = F(p++); L[l].cB  = F(p++);
    L[l].nhG = F(p++); L[l].nhB = F(p++);
    L[l].neG = F(p++); L[l].neB = F(p++);
    L[l].tW  = F(p++); L[l].tB  = F(p++);
    L[l].plG = F(p++); L[l].plB = F(p++);
    L[l].pW  = F(p++); L[l].pB  = F(p++);
  }
  const float *gnG = F(p++), *gnB = F(p++), *cvW = F(p++), *cvB = F(p++);

  // workspace carve-out (all sizes multiples of 256B)
  char* ws = (char*)d_ws;
  size_t off = 0;
  auto take = [&](size_t bytes) { void* r = ws + off;
                                  off += (bytes + 255) & ~size_t(255); return r; };
  float*    e    = (float*)take((size_t)NROW*HF*4);       // 134 MB, in-place e
  float*    h    = (float*)take((size_t)HROW*HF*4);
  float*    Uh   = (float*)take((size_t)HROW*HF*4);
  float*    Vh   = (float*)take((size_t)HROW*HF*4);
  float*    Ahb  = (float*)take((size_t)HROW*HF*4);
  float*    Bhb  = (float*)take((size_t)HROW*HF*4);
  float*    aggP = (float*)take((size_t)HROW*2*HF*4);
  float*    tvec = (float*)take((size_t)3*NB*HF*4);
  float*    gPS  = (float*)take((size_t)1024*32*4);
  float*    gPQ  = (float*)take((size_t)1024*32*4);
  _Float16* pack = (_Float16*)take((size_t)NMAT*MATH*2);

  PackArgs pa;
  pa.src[0] = eeW;
  for (int l = 0; l < 3; ++l) {
    pa.src[1 + 6*l] = L[l].cW;
    pa.src[2 + 6*l] = L[l].pW;
    pa.src[3 + 6*l] = L[l].uW;
    pa.src[4 + 6*l] = L[l].vW;
    pa.src[5 + 6*l] = L[l].aW;
    pa.src[6 + 6*l] = L[l].bW;
  }
  k_pack<<<(NMAT*MATH + 255)/256, 256, 0, stream>>>(pa, pack);
  k_h0<<<(HROW*HF)/256, 256, 0, stream>>>(x, h);
  TembArgs ta;
  ta.ts = ts; ta.t1W = t1W; ta.t1B = t1B; ta.t2W = t2W; ta.t2B = t2B;
  ta.tW0 = L[0].tW; ta.tB0 = L[0].tB;
  ta.tW1 = L[1].tW; ta.tB1 = L[1].tB;
  ta.tW2 = L[2].tW; ta.tB2 = L[2].tB;
  ta.tvec = tvec;
  k_temb<<<1, 128, 0, stream>>>(ta);
  k_embed<<<NROW/128, 256, 0, stream>>>(graph, pack + 0*MATH, eeB, e);

  for (int l = 0; l < 3; ++l) {
    k_nodes<<<8, 256, 0, stream>>>(h,
        pack + (3 + 6*l)*MATH, pack + (4 + 6*l)*MATH,
        pack + (5 + 6*l)*MATH, pack + (6 + 6*l)*MATH,
        L[l].uB, L[l].vB, L[l].aB, L[l].bB,
        Uh, Vh, Ahb, Bhb);
    k_edge<<<NROW/128, 256, 0, stream>>>(e, Ahb, Bhb, Vh,
        pack + (1 + 6*l)*MATH, pack + (2 + 6*l)*MATH,
        L[l].cB, L[l].pB, L[l].neG, L[l].neB, L[l].plG, L[l].plB,
        tvec + (size_t)l*NB*HF, aggP);
    k_hupdate<<<HROW, 128, 0, stream>>>(h, Uh, aggP, L[l].nhG, L[l].nhB);
  }
  k_gnstats<<<1024, 256, 0, stream>>>(e, gPS, gPQ);
  k_out<<<1024, 256, 0, stream>>>(e, gnG, gnB, cvW, cvB, gPS, gPQ, (float*)d_out);
}
